// HeteroLSTM_50766513439447
// MI455X (gfx1250) — compile-verified, matrix operand order checked by reading, not run on hardware
//
#include <hip/hip_runtime.h>
#include <hip/hip_bf16.h>

// ---------------------------------------------------------------------------
// HeteroLSTM on MI455X (gfx1250): bf16 WMMA GEMMs + fp32 pointwise recurrence
// B=256, S=16, N=64 agents, H=128.  Gate column order: 0=i,1=f,2=c,3=o.
// ---------------------------------------------------------------------------

typedef __bf16 bf16_t;
typedef __attribute__((ext_vector_type(16))) __bf16 v16bf;
typedef __attribute__((ext_vector_type(8)))  __bf16 v8bf;
typedef __attribute__((ext_vector_type(8)))  float  v8f;

#define HD 128
#define GH 512          // 4 gates * H
#define BB 256
#define NA 64
#define ST 16
#define BN (BB*NA)      // 16384

__device__ __forceinline__ bf16_t f2bf(float f) {
    unsigned u = __builtin_bit_cast(unsigned, f);
    unsigned r = u + 0x7FFFu + ((u >> 16) & 1u);     // round-to-nearest-even
    unsigned short s = (unsigned short)(r >> 16);
    return __builtin_bit_cast(bf16_t, s);
}

__device__ __forceinline__ float sigf(float x) { return 1.0f / (1.0f + __expf(-x)); }

// ---- parameter pointer tables (flat order of jax pytree: sorted dict keys) --
struct GateParams {
    const float* W[6];    // nodes sorted: agent, agent_summ, hideout, hideout_summ, state_summ, timestep
    const float* b[6];
    const float* Wl[10];  // edges sorted: a_to_as, as_rev_a, as_to_ss, h_to_hs, hs_rev_h,
    const float* Wr[10];  //               hs_to_ss, ss_rev_as, ss_rev_hs, ss_rev_ts, ts_to_ss
    const float* bl[10];
    const float* wc[6];   // null for gate 'c'
};
struct AllParams { GateParams g[4]; };   // order i, f, c, o

struct GemmDesc { const bf16_t* A; const bf16_t* W; float* out;
                  const float* add; const float* bias; int K; };
struct Gemm6 { GemmDesc d[6]; };
struct TsX    { const float* W[4]; };
struct AgentX { const float* Wx[4]; };

// ---------------------------------------------------------------------------
// Pack stacked weights into bf16 WMMA-B tiles.
// Tile (kt,nt) of problem p lives at base_p + (kt*32+nt)*512 + lane*16 + e.
// Lane/elem -> (K,N):  Nloc = lane&15;  K = kt*32 + (e<8?e:e+8) + (lane>=16?8:0)
// Problems: 0 state_summ(K=4H) 1 agent_summ(3H) 2 hideout_summ(3H)
//           3 hideout(2H) 4 timestep(2H) 5 agent_Wl(H) 6 agent_Wr(H)
// ---------------------------------------------------------------------------
__global__ void k_pack(AllParams P, bf16_t* wpack) {
    int id = blockIdx.x * 256 + threadIdx.x;
    if (id >= 16 * HD * GH) return;
    const int ks[7] = {4*HD, 3*HD, 3*HD, 2*HD, 2*HD, HD, HD};
    int p = 0, rel = id;
    for (; p < 7; ++p) { int sz = ks[p] * GH; if (rel < sz) break; rel -= sz; }
    int tile = rel >> 9;
    int idx  = rel & 511;
    int lane = idx >> 4, e = idx & 15;
    int nt = tile & 31, kt = tile >> 5;
    int K   = kt * 32 + ((e < 8) ? e : e + 8) + ((lane >= 16) ? 8 : 0);
    int col = nt * 16 + (lane & 15);
    int g = col >> 7, j = col & 127;
    const GateParams& gp = P.g[g];
    int blk = K >> 7, o = (K & 127) * HD + j;
    float v;
    switch (p) {
      case 0: v = (blk==0) ? gp.Wl[5][o] : (blk==1) ? gp.Wl[2][o] : (blk==2) ? gp.Wl[9][o]
                           : (gp.Wr[5][o] + gp.Wr[2][o] + gp.Wr[9][o]); break;
      case 1: v = (blk==0) ? gp.Wl[0][o] : (blk==1) ? gp.Wl[6][o]
                           : (gp.Wr[0][o] + gp.Wr[6][o]); break;
      case 2: v = (blk==0) ? gp.Wl[3][o] : (blk==1) ? gp.Wl[7][o]
                           : (gp.Wr[3][o] + gp.Wr[7][o]); break;
      case 3: v = (blk==0) ? gp.Wl[4][o] : gp.Wr[4][o]; break;
      case 4: v = (blk==0) ? gp.Wl[8][o] : gp.Wr[8][o]; break;
      case 5: v = gp.Wl[1][o]; break;
      default: v = gp.Wr[1][o]; break;
    }
    wpack[id] = f2bf(v);
}

// ---- combined biases (node bias + sage bl's) and step-invariant hideout x ---
__global__ void k_misc(AllParams P, float* bias, float* xh, const float* ho) {
    int id = blockIdx.x * 256 + threadIdx.x;
    if (id < 6 * GH) {
        int t = id >> 9, c = id & 511, g = c >> 7, j = c & 127;
        const GateParams& gp = P.g[g];
        float v;
        switch (t) {
          case 0: v = gp.b[4][j] + gp.bl[5][j] + gp.bl[2][j] + gp.bl[9][j]; break; // state_summ
          case 1: v = gp.b[1][j] + gp.bl[0][j] + gp.bl[6][j]; break;               // agent_summ
          case 2: v = gp.b[3][j] + gp.bl[3][j] + gp.bl[7][j]; break;               // hideout_summ
          case 3: v = gp.b[2][j] + gp.bl[4][j]; break;                             // hideout
          case 4: v = gp.b[5][j] + gp.bl[8][j]; break;                             // timestep
          default: v = gp.b[0][j] + gp.bl[1][j]; break;                            // agent
        }
        bias[id] = v;
    } else {
        int i = id - 6 * GH;
        if (i < BB * GH) {
            int b = i >> 9, c = i & 511, g = c >> 7, j = c & 127;
            xh[i] = ho[b*2] * P.g[g].W[2][j] + ho[b*2+1] * P.g[g].W[2][HD + j];
        }
    }
}

__global__ void k_zero(float* p, int n) {
    int id = blockIdx.x * 256 + threadIdx.x;
    if (id < n) p[id] = 0.0f;
}

// ---------------------------------------------------------------------------
// Per-step operand prep: build bf16 A matrices for all GEMMs + timestep x term.
// h_small layout: t*32768 with t: 0=ss 1=as 2=hs 3=h 4=ts
// ---------------------------------------------------------------------------
__global__ void k_prep(int s, const float* hsml, const float* h_agent,
                       const float* ts_obs, TsX tx,
                       bf16_t* a_small, bf16_t* a_agent, float* xts, bf16_t* a_bcast) {
    int id = blockIdx.x * 256 + threadIdx.x;
    if (id < 458752) {
        float v;
        int i = id;
        if (i < 131072) {                                   // A_ss: [h_hs,h_as,h_ts,h_ss]
            int b = i >> 9, c = i & 511, jb = c >> 7, j = c & 127;
            const int t4[4] = {2, 1, 4, 0};
            v = hsml[t4[jb] * 32768 + b * 128 + j];
        } else if (i < 229376) {                            // A_as: [mean_a,h_ss,h_as]
            int r = i - 131072, b = r / 384, c = r % 384, jb = c >> 7, j = c & 127;
            if (jb == 0) {
                float sum = 0.0f;
                const float* hp = h_agent + b * 8192 + j;
                #pragma unroll 8
                for (int n = 0; n < 64; ++n) sum += hp[n * 128];
                v = sum * (1.0f / 64.0f);
            } else v = hsml[((jb == 1) ? 0 : 1) * 32768 + b * 128 + j];
        } else if (i < 327680) {                            // A_hs: [h_h,h_ss,h_hs]
            int r = i - 229376, b = r / 384, c = r % 384, jb = c >> 7, j = c & 127;
            const int t3[3] = {3, 0, 2};
            v = hsml[t3[jb] * 32768 + b * 128 + j];
        } else if (i < 393216) {                            // A_h: [h_hs,h_h]
            int r = i - 327680, b = r >> 8, c = r & 255, jb = c >> 7, j = c & 127;
            v = hsml[((jb == 0) ? 2 : 3) * 32768 + b * 128 + j];
        } else {                                            // A_ts: [h_ss,h_ts]
            int r = i - 393216, b = r >> 8, c = r & 255, jb = c >> 7, j = c & 127;
            v = hsml[((jb == 0) ? 0 : 4) * 32768 + b * 128 + j];
        }
        a_small[id] = f2bf(v);
    } else if (id < 458752 + 2097152) {                     // A_agent = bf16(h_agent)
        int i = id - 458752;
        a_agent[i] = f2bf(h_agent[i]);
    } else if (id < 458752 + 2097152 + 131072) {            // xts = ts * W_timestep
        int i = id - 458752 - 2097152;
        int b = i >> 9, c = i & 511, g = c >> 7, j = c & 127;
        xts[i] = ts_obs[b * ST + s] * tx.W[g][j];
    } else if (id < 458752 + 2097152 + 131072 + 32768) {    // A_bcast = bf16(h_as)
        int i = id - 458752 - 2097152 - 131072;
        a_bcast[i] = f2bf(hsml[32768 + i]);
    }
}

// ---------------------------------------------------------------------------
// WMMA GEMM, one wave per 16x16 output tile. M=256, N=512, K per problem.
// ---------------------------------------------------------------------------
__global__ void k_gemm_small(Gemm6 gb) {
    GemmDesc de = gb.d[blockIdx.z];
    int wave = threadIdx.x >> 5, l = threadIdx.x & 31;
    int ct = blockIdx.x * 4 + wave;       // 0..31 column tiles
    int rt = blockIdx.y;                  // 0..15 row tiles
    int K  = de.K;
    v8f acc = {};
    const bf16_t* Arow = de.A + (rt * 16 + (l & 15)) * K + ((l >> 4) << 3);
    const bf16_t* Wp   = de.W + ct * 512 + l * 16;
    for (int kt = 0; kt < (K >> 5); ++kt) {
        __builtin_prefetch(Wp + 32 * 512, 0, 1);
        v8bf lo = *(const v8bf*)(Arow);
        v8bf hi = *(const v8bf*)(Arow + 16);
        v16bf a = __builtin_shufflevector(lo, hi, 0,1,2,3,4,5,6,7,8,9,10,11,12,13,14,15);
        v16bf b = *(const v16bf*)(Wp);
        acc = __builtin_amdgcn_wmma_f32_16x16x32_bf16(false, a, false, b,
                                                      (short)0, acc, false, false);
        Arow += 32; Wp += 32 * 512;
    }
    int col = ct * 16 + (l & 15);
    int rbase = rt * 16 + ((l >> 4) << 3);
    float bi = de.bias ? de.bias[col] : 0.0f;
    #pragma unroll
    for (int v = 0; v < 8; ++v) {
        int row = rbase + v;
        float r = acc[v] + bi;
        if (de.add) r += de.add[row * GH + col];
        de.out[row * GH + col] = r;
    }
}

// Big agent GEMM: [16384,128] x [128,512]; epilogue adds bias + per-batch
// broadcast (agent_summ@Wl) + input projection agent_obs@W_agent (K=8).
__global__ void k_gemm_agent(const bf16_t* A, const bf16_t* W, float* out,
                             const float* bcast, const float* bias,
                             const float* agent_obs, AgentX ax, int s) {
    int wave = threadIdx.x >> 5, l = threadIdx.x & 31;
    int ct = blockIdx.x * 4 + wave;       // 0..31
    int rt = blockIdx.y;                  // 0..1023
    v8f acc = {};
    const bf16_t* Arow = A + (rt * 16 + (l & 15)) * HD + ((l >> 4) << 3);
    const bf16_t* Wp   = W + ct * 512 + l * 16;
    #pragma unroll
    for (int kt = 0; kt < 4; ++kt) {
        v8bf lo = *(const v8bf*)(Arow);
        v8bf hi = *(const v8bf*)(Arow + 16);
        v16bf a = __builtin_shufflevector(lo, hi, 0,1,2,3,4,5,6,7,8,9,10,11,12,13,14,15);
        v16bf b = *(const v16bf*)(Wp);
        acc = __builtin_amdgcn_wmma_f32_16x16x32_bf16(false, a, false, b,
                                                      (short)0, acc, false, false);
        Arow += 32; Wp += 32 * 512;
    }
    int col = ct * 16 + (l & 15);
    int g = col >> 7, j = col & 127;
    float wv[8];
    #pragma unroll
    for (int k = 0; k < 8; ++k) wv[k] = ax.Wx[g][k * HD + j];
    float bi = bias[col];
    int rbase = rt * 16 + ((l >> 4) << 3);
    #pragma unroll
    for (int v = 0; v < 8; ++v) {
        int row = rbase + v;
        int b = row >> 6, n = row & 63;
        const float* ob = agent_obs + (((b * ST + s) * NA + n) << 3);
        float xs = 0.0f;
        #pragma unroll
        for (int k = 0; k < 8; ++k) xs += wv[k] * ob[k];
        out[row * GH + col] = acc[v] + bi + bcast[b * GH + col] + xs;
    }
}

// ---------------------------------------------------------------------------
// Pointwise LSTM updates (fp32, peepholes: i/f on c_prev, o on c_new)
// ---------------------------------------------------------------------------
__global__ void k_point_small(AllParams P, const float* pre, float* h, float* c,
                              float* outp, int last) {
    int id = blockIdx.x * 256 + threadIdx.x;
    if (id >= 5 * BB * HD) return;
    int t = id >> 15;                     // node slot 0=ss 1=as 2=hs 3=h 4=ts
    int r = id & 32767, b = r >> 7, j = r & 127;
    const int nodeIdx[5] = {4, 1, 3, 2, 5};
    int nd = nodeIdx[t];
    const float* pr = pre + t * (BB * GH) + b * GH;
    float cc = c[id];
    float pi = pr[j]        + P.g[0].wc[nd][j] * cc;
    float pf = pr[HD + j]   + P.g[1].wc[nd][j] * cc;
    float gv = tanhf(pr[2 * HD + j]);
    float iv = sigf(pi), fv = sigf(pf);
    float cn = fv * cc + iv * gv;
    float po = pr[3 * HD + j] + P.g[3].wc[nd][j] * cn;
    float hn = sigf(po) * tanhf(cn);
    c[id] = cn; h[id] = hn;
    if (last && t == 0) outp[r] = hn;     // h['state_summ'] -> d_out
}

__global__ void k_point_agent(AllParams P, const float* pre, float* h, float* c) {
    int id = blockIdx.x * 256 + threadIdx.x;
    if (id >= BN * HD) return;
    int row = id >> 7, j = id & 127;
    const float* pr = pre + row * GH;
    float cc = c[id];
    float pi = pr[j]        + P.g[0].wc[0][j] * cc;
    float pf = pr[HD + j]   + P.g[1].wc[0][j] * cc;
    float gv = tanhf(pr[2 * HD + j]);
    float iv = sigf(pi), fv = sigf(pf);
    float cn = fv * cc + iv * gv;
    float po = pr[3 * HD + j] + P.g[3].wc[0][j] * cn;
    c[id] = cn; h[id] = sigf(po) * tanhf(cn);
}

// ---------------------------------------------------------------------------
static void fill_gate(GateParams& gp, void* const* d_in, int base, bool has_wc) {
    for (int n = 0; n < 6; ++n) gp.W[n] = (const float*)d_in[base + n];
    for (int n = 0; n < 6; ++n) gp.b[n] = (const float*)d_in[base + 6 + n];
    for (int e = 0; e < 10; ++e) {
        gp.Wl[e] = (const float*)d_in[base + 12 + 3 * e];
        gp.Wr[e] = (const float*)d_in[base + 12 + 3 * e + 1];
        gp.bl[e] = (const float*)d_in[base + 12 + 3 * e + 2];
    }
    for (int n = 0; n < 6; ++n) gp.wc[n] = has_wc ? (const float*)d_in[base + 42 + n] : nullptr;
}

extern "C" void kernel_launch(void* const* d_in, const int* in_sizes, int n_in,
                              void* d_out, int out_size, void* d_ws, size_t ws_size,
                              hipStream_t stream) {
    (void)in_sizes; (void)n_in; (void)out_size; (void)ws_size;
    const float* agent_obs   = (const float*)d_in[0];
    const float* hideout_obs = (const float*)d_in[1];
    const float* ts_obs      = (const float*)d_in[2];
    // d_in[3] = num_agents (constant N) unused.
    // params pytree (sorted keys): gates c(4..45), f(46..93), i(94..141), o(142..189)
    AllParams P;
    fill_gate(P.g[0], d_in, 94, true);    // i
    fill_gate(P.g[1], d_in, 46, true);    // f
    fill_gate(P.g[2], d_in, 4,  false);   // c
    fill_gate(P.g[3], d_in, 142, true);   // o

    char* ws = (char*)d_ws;
    size_t off = 0;
    auto carve = [&](size_t bytes) -> char* {
        char* p = ws + off;
        off = (off + bytes + 255) & ~(size_t)255;
        return p;
    };
    bf16_t* wpack   = (bf16_t*)carve((size_t)16 * HD * GH * 2);     // 2 MB packed weights
    bf16_t* a_small = (bf16_t*)carve((size_t)458752 * 2);
    bf16_t* a_agent = (bf16_t*)carve((size_t)BN * HD * 2);
    bf16_t* a_bcast = (bf16_t*)carve((size_t)BB * HD * 2);
    float*  states  = (float*)carve((size_t)(163840 * 2 + 2097152 * 2) * 4);
    float*  h_small = states;
    float*  c_small = states + 163840;
    float*  h_agent = states + 327680;
    float*  c_agent = states + 327680 + 2097152;
    float*  pre_small = (float*)carve((size_t)6 * BB * GH * 4);
    float*  pre_agent = (float*)carve((size_t)BN * GH * 4);
    float*  xh   = (float*)carve((size_t)BB * GH * 4);
    float*  xts  = (float*)carve((size_t)BB * GH * 4);
    float*  bias = (float*)carve((size_t)6 * GH * 4);

    k_pack<<<4096, 256, 0, stream>>>(P, wpack);
    k_misc<<<(6 * GH + BB * GH + 255) / 256, 256, 0, stream>>>(P, bias, xh, hideout_obs);
    k_zero<<<17664, 256, 0, stream>>>(states, 4521984);

    const int U = HD * GH;   // 65536 elements per H-row-block of packed weights
    Gemm6 gb;
    gb.d[0] = { a_small + 0,      wpack + 0  * U, pre_small + 0 * 131072, nullptr, bias + 0 * 512, 512 };
    gb.d[1] = { a_small + 131072, wpack + 4  * U, pre_small + 1 * 131072, nullptr, bias + 1 * 512, 384 };
    gb.d[2] = { a_small + 229376, wpack + 7  * U, pre_small + 2 * 131072, nullptr, bias + 2 * 512, 384 };
    gb.d[3] = { a_small + 327680, wpack + 10 * U, pre_small + 3 * 131072, xh,      bias + 3 * 512, 256 };
    gb.d[4] = { a_small + 393216, wpack + 12 * U, pre_small + 4 * 131072, xts,     bias + 4 * 512, 256 };
    gb.d[5] = { a_bcast,          wpack + 14 * U, pre_small + 5 * 131072, nullptr, nullptr,        128 };

    TsX    tx = {{ P.g[0].W[5], P.g[1].W[5], P.g[2].W[5], P.g[3].W[5] }};
    AgentX ax = {{ P.g[0].W[0], P.g[1].W[0], P.g[2].W[0], P.g[3].W[0] }};

    for (int s = 0; s < ST; ++s) {
        k_prep<<<10624, 256, 0, stream>>>(s, h_small, h_agent, ts_obs, tx,
                                          a_small, a_agent, xts, a_bcast);
        k_gemm_small<<<dim3(8, 16, 6), 128, 0, stream>>>(gb);
        k_gemm_agent<<<dim3(8, 1024), 128, 0, stream>>>(a_agent, wpack + 15 * U, pre_agent,
                                                        pre_small + 5 * 131072, bias + 5 * 512,
                                                        agent_obs, ax, s);
        k_point_small<<<640, 256, 0, stream>>>(P, pre_small, h_small, c_small,
                                               (float*)d_out, s == ST - 1);
        k_point_agent<<<8192, 256, 0, stream>>>(P, pre_agent, h_agent, c_agent);
    }
}
